// HeteroGraphConv_74612171866592
// MI455X (gfx1250) — matrix operand order repaired and checked by reading, not compiled, not run on hardware
//
#include <hip/hip_runtime.h>
#include <hip/hip_bf16.h>

// ---------------------------------------------------------------------------
// CDNA5 (gfx1250) implementation of the 2-layer HeteroConv GNN + pool + MLP.
// GEMMs: v_wmma_f32_16x16x32_bf16 (wave32), A operand DMA'd to LDS by the
// Tensor Data Mover (tensor_load_to_lds + s_wait_tensorcnt, double-buffered),
// B operand staged transposed in LDS as bf16 for vector fragment loads.
// Edge aggregation is reassociated:  segsum(x[src]) @ W == segsum((x@W)[src])
// so we project to H=64 first and scatter narrow rows with f32 atomics.
// ---------------------------------------------------------------------------

typedef __attribute__((ext_vector_type(16))) __bf16 v16bf;
typedef __attribute__((ext_vector_type(8)))  float  v8f;
typedef __attribute__((ext_vector_type(4)))  unsigned int v4u;
typedef __attribute__((ext_vector_type(8)))  int v8i_;
typedef __attribute__((ext_vector_type(4)))  int v4i_;

// Issue a TDM load of a 16-row x 32-col f32 tile (row stride = strideElems)
// from global memory into LDS at byte offset ldsOff.  D# per ISA ch.8:
//  group0: count=1 | lds_addr | global_addr[56:0] | type=2
//  group1: data_size=2(4B), tensor_dim0=32, tensor_dim1=16,
//          tile_dim0=32, tile_dim1=16, tensor_dim0_stride=strideElems
__device__ __forceinline__ void tdm_load_tile16x32(unsigned ldsOff,
                                                   const float* gptr,
                                                   int strideElems) {
  unsigned long long ga = (unsigned long long)(uintptr_t)gptr;
  v4u g0;
  g0[0] = 1u;                                   // count=1 (valid user D#)
  g0[1] = ldsOff;                               // lds_addr (bytes)
  g0[2] = (unsigned)(ga & 0xFFFFFFFFu);         // global_addr[31:0]
  g0[3] = (unsigned)((ga >> 32) & 0x01FFFFFFu)  // global_addr[56:32]
          | (2u << 30);                         // type=2 ("image")
  v8i_ g1;
  g1[0] = (int)(2u << 16);                      // wg_mask=0, data_size=4B
  g1[1] = (int)(32u << 16);                     // tensor_dim0[15:0]=32 @bit48
  g1[2] = (int)(16u << 16);                     // tensor_dim1[15:0]=16 @bit80
  g1[3] = (int)(32u << 16);                     // tile_dim0=32 @bit112
  g1[4] = 16;                                   // tile_dim1=16, tile_dim2=0
  g1[5] = strideElems;                          // tensor_dim0_stride[31:0]
  g1[6] = 0;                                    // stride hi / dim1_stride lo
  g1[7] = 0;
  v4i_ z4 = {0, 0, 0, 0};
#if defined(__clang_major__) && (__clang_major__ >= 23)
  v8i_ z8 = {0, 0, 0, 0, 0, 0, 0, 0};
  __builtin_amdgcn_tensor_load_to_lds(g0, g1, z4, z4, z8, 0);
#else
  __builtin_amdgcn_tensor_load_to_lds(g0, g1, z4, z4, 0);
#endif
}

// C[M,64] = A[M,KD] @ W[KD,64]   (bf16 WMMA, f32 accumulate)
// EPI: C[idx] = leaky_relu(C[idx] + gemm + bias[n])   (in-place epilogue)
template<int KD, bool EPI>
__global__ __launch_bounds__(128)
void gemm_wmma_kernel(const float* __restrict__ A, const float* __restrict__ W,
                      const float* __restrict__ bias, float* __restrict__ C,
                      int M) {
  constexpr int KSTEPS = KD / 32;
  __shared__ __attribute__((aligned(32))) __bf16 wldsT[64 * KD];   // [n][k]
  __shared__ __attribute__((aligned(16))) float  aldsf[4][2][16 * 32];

  const int tid = threadIdx.x;
  // Stage weight panel transposed as bf16 (native cvt).
  for (int i = tid; i < KD * 64; i += 128) {
    int k = i >> 6, n = i & 63;
    wldsT[n * KD + k] = (__bf16)W[i];
  }
  __syncthreads();

  const int wave = tid >> 5, lane = tid & 31;
  const int lrow = lane & 15, half = lane >> 4;
  const long rowbase = (long)blockIdx.x * 64 + (long)wave * 16;
  if (rowbase >= M) return;               // all M are multiples of 16

  const float* atile = A + (size_t)rowbase * KD;
  unsigned ldsOff[2];
  ldsOff[0] = (unsigned)(uintptr_t)&aldsf[wave][0][0];  // flat->LDS: addr[31:0]
  ldsOff[1] = (unsigned)(uintptr_t)&aldsf[wave][1][0];

  // Prologue: DMA first A tile, then double-buffer.
  tdm_load_tile16x32(ldsOff[0], atile, KD);

  v8f acc[4];
  #pragma unroll
  for (int nt = 0; nt < 4; ++nt) acc[nt] = 0.0f;

  #pragma unroll
  for (int ks = 0; ks < KSTEPS; ++ks) {
    const int cur = ks & 1;
    if (ks + 1 < KSTEPS)
      tdm_load_tile16x32(ldsOff[cur ^ 1], atile + (ks + 1) * 32, KD);
    if (ks + 1 < KSTEPS) __builtin_amdgcn_s_wait_tensorcnt((short)1);
    else                 __builtin_amdgcn_s_wait_tensorcnt((short)0);
    asm volatile("" ::: "memory");        // keep LDS reads below the wait

    const float* as = &aldsf[wave][cur][0];
    v16bf af;   // A frag: element s -> K = 16*half + s (matches B frag)
    #pragma unroll
    for (int j = 0; j < 8; ++j) {
      const int base = 2 * j + 16 * half;
      float2 p = *(const float2*)&as[lrow * 32 + base];
      af[2 * j]     = (__bf16)p.x;
      af[2 * j + 1] = (__bf16)p.y;
    }
    #pragma unroll
    for (int nt = 0; nt < 4; ++nt) {
      // B frag: lanes 0-15 K=ks*32..+15, lanes 16-31 K=+16..+31, col=lrow.
      const v16bf bf =
          *(const v16bf*)&wldsT[(nt * 16 + lrow) * KD + ks * 32 + 16 * half];
      acc[nt] = __builtin_amdgcn_wmma_f32_16x16x32_bf16(
          false, af, false, bf, (short)0, acc[nt], false, false);
    }
  }

  // C/D layout: VGPR r -> row = half*8 + r, col = lrow (per 16x16 tile)
  #pragma unroll
  for (int nt = 0; nt < 4; ++nt) {
    #pragma unroll
    for (int r = 0; r < 8; ++r) {
      size_t row = (size_t)rowbase + half * 8 + r;
      size_t idx = row * 64 + nt * 16 + lrow;
      float v = acc[nt][r];
      if (EPI) {
        v += C[idx] + bias[nt * 16 + lrow];
        v = v > 0.0f ? v : 0.01f * v;     // leaky_relu, slope 0.01
      }
      C[idx] = v;
    }
  }
}

// acc[dst] += P[src]  for each edge; 32 lanes per edge, 2 floats per lane.
__global__ __launch_bounds__(256)
void scatter_edges_kernel(const float* __restrict__ P,
                          const int* __restrict__ edges, int E,
                          float* __restrict__ acc) {
  long gid = (long)blockIdx.x * blockDim.x + threadIdx.x;
  long e = gid >> 5;
  int lane = gid & 31;
  if (e >= E) return;
  __builtin_prefetch(edges + e + 256, 0, 0);          // global_prefetch_b8
  int s = edges[e];
  int d = edges[(long)E + e];
  float v0 = P[(size_t)s * 64 + lane];
  float v1 = P[(size_t)s * 64 + 32 + lane];
  atomicAdd(&acc[(size_t)d * 64 + lane], v0);
  atomicAdd(&acc[(size_t)d * 64 + 32 + lane], v1);
}

__global__ void fill_zero_kernel(float* __restrict__ p, size_t n) {
  size_t i = (size_t)blockIdx.x * blockDim.x + threadIdx.x;
  if (i < n) p[i] = 0.0f;
}

// out = a + (b?b:0) + (c?c:0)   (summing W_root / b_rel across incoming rels)
__global__ void sum3_kernel(const float* __restrict__ a,
                            const float* __restrict__ b,
                            const float* __restrict__ c,
                            float* __restrict__ out, int n) {
  int i = blockIdx.x * blockDim.x + threadIdx.x;
  if (i >= n) return;
  float v = a[i];
  if (b) v += b[i];
  if (c) v += c[i];
  out[i] = v;
}

// mean-pool accumulation: psum[g][c] += x[node][c]; pcnt[g] += 1
__global__ __launch_bounds__(256)
void pool_accum_kernel(const float* __restrict__ x,
                       const int* __restrict__ batch, int n,
                       float* __restrict__ psum, float* __restrict__ pcnt) {
  long gid = (long)blockIdx.x * blockDim.x + threadIdx.x;
  long node = gid >> 6;
  int c = gid & 63;
  if (node >= n) return;
  int g = batch[node];
  atomicAdd(&psum[(size_t)g * 64 + c], x[(size_t)node * 64 + c]);
  if (c == 0) atomicAdd(&pcnt[g], 1.0f);
}

// per-graph head: concat(5 pooled means, post_emb) -> lin1 -> lin2 -> softmax
__global__ __launch_bounds__(64)
void head_kernel(const float* __restrict__ psum,   // [5][G][64]
                 const float* __restrict__ pcnt,   // [5][G]
                 const float* __restrict__ post,   // [G][768]
                 const float* __restrict__ W1, const float* __restrict__ b1,
                 const float* __restrict__ W2, const float* __restrict__ b2,
                 float* __restrict__ out, int G) {
  const int g = blockIdx.x, t = threadIdx.x;   // 64 threads
  __shared__ float f[1088];
  __shared__ float h[64];
  for (int i = t; i < 320; i += 64) {
    int ty = i >> 6, c = i & 63;
    float cnt = pcnt[(size_t)ty * G + g];
    f[i] = psum[((size_t)ty * G + g) * 64 + c] / fmaxf(cnt, 1.0f);
  }
  for (int i = 320 + t; i < 1088; i += 64) f[i] = post[(size_t)g * 768 + (i - 320)];
  __syncthreads();
  float a = b1[t];
  for (int i = 0; i < 1088; ++i) a += f[i] * W1[(size_t)i * 64 + t];
  h[t] = a > 0.0f ? a : 0.01f * a;
  __syncthreads();
  if (t == 0) {
    float o0 = b2[0], o1 = b2[1];
    for (int k = 0; k < 64; ++k) { o0 += h[k] * W2[k * 2]; o1 += h[k] * W2[k * 2 + 1]; }
    o0 = o0 > 0.0f ? o0 : 0.01f * o0;
    o1 = o1 > 0.0f ? o1 : 0.01f * o1;
    float m = fmaxf(o0, o1);
    float e0 = __expf(o0 - m), e1 = __expf(o1 - m);
    float s = e0 + e1;
    out[(size_t)g * 2] = e0 / s;
    out[(size_t)g * 2 + 1] = e1 / s;
  }
}

// ---------------------------------------------------------------------------
// Host orchestration
// ---------------------------------------------------------------------------
namespace {
struct RelInfo { int s, d; };
// RELS order: (tag,q)(tag,a)(tag,c)(mod,q)(mod,a)(q,tag)(a,tag)(c,tag)(q,mod)(a,mod)
const RelInfo g_rels[10] = {{3,0},{3,1},{3,2},{4,0},{4,1},{0,3},{1,3},{2,3},{0,4},{1,4}};
// incoming relation indices per destination type (-1 = none)
const int g_in[5][3] = {
  {0, 3, -1},   // question <- e_tag_q, e_mod_q
  {1, 4, -1},   // answer   <- e_tag_a, e_mod_a
  {2, -1, -1},  // comment  <- e_tag_c
  {5, 6, 7},    // tag      <- e_q_tag, e_a_tag, e_c_tag
  {8, 9, -1},   // module   <- e_q_mod, e_a_mod
};
}  // namespace

extern "C" void kernel_launch(void* const* d_in, const int* in_sizes, int n_in,
                              void* d_out, int out_size, void* d_ws, size_t ws_size,
                              hipStream_t stream) {
  (void)n_in; (void)ws_size;
  // ---- input mapping (setup_inputs() dict insertion order) ----
  const float* X0[5];
  int Nn[5], Dd0[5];
  for (int t = 0; t < 5; ++t) {
    Nn[t]  = in_sizes[15 + t];                   // b_<type> has N[t] elements
    X0[t]  = (const float*)d_in[t];
    Dd0[t] = in_sizes[t] / Nn[t];                // 256 or 64
  }
  const int E = in_sizes[5] / 2;
  const int G = in_sizes[20] / 768;
  // params: idx 21.. : per layer, per relation: W_rel, b_rel, W_root
  auto Wrel  = [&](int l, int r) { return (const float*)d_in[21 + l * 30 + r * 3 + 0]; };
  auto brel  = [&](int l, int r) { return (const float*)d_in[21 + l * 30 + r * 3 + 1]; };
  auto Wroot = [&](int l, int r) { return (const float*)d_in[21 + l * 30 + r * 3 + 2]; };
  const float* lin1_W = (const float*)d_in[81];
  const float* lin1_b = (const float*)d_in[82];
  const float* lin2_W = (const float*)d_in[83];
  const float* lin2_b = (const float*)d_in[84];

  // ---- workspace carve-up ----
  float* ws = (float*)d_ws;
  size_t off = 0;
  float* accA[5]; for (int t = 0; t < 5; ++t) { accA[t] = ws + off; off += (size_t)Nn[t] * 64; }
  float* accB[5]; for (int t = 0; t < 5; ++t) { accB[t] = ws + off; off += (size_t)Nn[t] * 64; }
  int maxN = 0; for (int t = 0; t < 5; ++t) maxN = Nn[t] > maxN ? Nn[t] : maxN;
  float* proj = ws + off; off += (size_t)maxN * 64;
  float* wsum = ws + off; off += 256 * 64;
  float* bsum = ws + off; off += 64;
  float* psum = ws + off; off += (size_t)5 * G * 64;
  float* pcnt = ws + off; off += (size_t)5 * G;

  auto fill0 = [&](float* p, size_t n) {
    fill_zero_kernel<<<(unsigned)((n + 255) / 256), 256, 0, stream>>>(p, n);
  };
  auto gemm = [&](const float* A, const float* W, const float* bias, float* C,
                  int M, int KD, bool epi) {
    dim3 grid((M + 63) / 64), block(128);
    if (KD == 256) {
      if (epi) gemm_wmma_kernel<256, true ><<<grid, block, 0, stream>>>(A, W, bias, C, M);
      else     gemm_wmma_kernel<256, false><<<grid, block, 0, stream>>>(A, W, bias, C, M);
    } else {
      if (epi) gemm_wmma_kernel<64,  true ><<<grid, block, 0, stream>>>(A, W, bias, C, M);
      else     gemm_wmma_kernel<64,  false><<<grid, block, 0, stream>>>(A, W, bias, C, M);
    }
  };

  // ---- two HeteroConv layers ----
  for (int l = 0; l < 2; ++l) {
    const float* curX[5];
    int curD[5];
    float** acc = (l == 0) ? accA : accB;
    for (int t = 0; t < 5; ++t) {
      curX[t] = (l == 0) ? X0[t] : accA[t];
      curD[t] = (l == 0) ? Dd0[t] : 64;
      fill0(acc[t], (size_t)Nn[t] * 64);
    }
    // per-relation: project source features, scatter-add into destination acc
    for (int r = 0; r < 10; ++r) {
      int s = g_rels[r].s, d = g_rels[r].d;
      gemm(curX[s], Wrel(l, r), nullptr, proj, Nn[s], curD[s], false);
      unsigned blocks = (unsigned)(((long)E * 32 + 255) / 256);
      scatter_edges_kernel<<<blocks, 256, 0, stream>>>(
          proj, (const int*)d_in[5 + r], E, acc[d]);
    }
    // per destination type: summed-root GEMM + summed bias + leaky_relu
    for (int t = 0; t < 5; ++t) {
      int r0 = g_in[t][0], r1 = g_in[t][1], r2 = g_in[t][2];
      const float* w1 = (r1 >= 0) ? Wroot(l, r1) : nullptr;
      const float* w2 = (r2 >= 0) ? Wroot(l, r2) : nullptr;
      const float* bb1 = (r1 >= 0) ? brel(l, r1) : nullptr;
      const float* bb2 = (r2 >= 0) ? brel(l, r2) : nullptr;
      int nw = curD[t] * 64;
      sum3_kernel<<<(nw + 255) / 256, 256, 0, stream>>>(Wroot(l, r0), w1, w2, wsum, nw);
      sum3_kernel<<<1, 64, 0, stream>>>(brel(l, r0), bb1, bb2, bsum, 64);
      gemm(curX[t], wsum, bsum, acc[t], Nn[t], curD[t], true);
    }
  }

  // ---- mean pool + MLP head + softmax ----
  fill0(psum, (size_t)5 * G * 64 + (size_t)5 * G);  // psum and pcnt contiguous
  for (int t = 0; t < 5; ++t) {
    unsigned blocks = (unsigned)(((long)Nn[t] * 64 + 255) / 256);
    pool_accum_kernel<<<blocks, 256, 0, stream>>>(
        accB[t], (const int*)d_in[15 + t], Nn[t],
        psum + (size_t)t * G * 64, pcnt + (size_t)t * G);
  }
  head_kernel<<<G, 64, 0, stream>>>(psum, pcnt, (const float*)d_in[20],
                                    lin1_W, lin1_b, lin2_W, lin2_b,
                                    (float*)d_out, G);
  (void)out_size;
}
